// MultiheadSelfAttention_66692252172315
// MI455X (gfx1250) — matrix operand order repaired and checked by reading, not compiled, not run on hardware
//
#include <hip/hip_runtime.h>

// MultiheadSelfAttention for MI455X (gfx1250), wave32 + WMMA.
// B=2, S=2048, D=1024, H=16, HD=64.
// Pipeline:
//   0) one-time f32->f16 conversion of weights + inputs (removes per-tile cvt)
//   1) qkv projections as f16 WMMA NT-GEMMs (V written transposed per head)
//   2) causal flash attention (online softmax, f16 WMMA, f32 accumulators)
//   3) output projection -> f32
// Workspace requirement: 28M _Float16 = 56 MB.

#define B_  2
#define S_  2048
#define D_  1024
#define H_  16
#define HD_ 64

typedef __attribute__((ext_vector_type(16))) _Float16 v16h;
typedef __attribute__((ext_vector_type(8)))  _Float16 v8h;
typedef __attribute__((ext_vector_type(8)))  float    v8f;
typedef __attribute__((ext_vector_type(4)))  float    v4f;

// ---- WMMA wrapper: D = A(16x32 f16) * B(32x16 f16) + C(16x16 f32) ----
__device__ __forceinline__ v8f wmma16(v16h a, v16h b, v8f c) {
    return __builtin_amdgcn_wmma_f32_16x16x32_f16(
        /*neg_a=*/false, a, /*neg_b=*/false, b,
        /*c_mod=*/(short)0, c, /*reuse_a=*/false, /*reuse_b=*/false);
}

// ---- A-operand loader (lane = M mod 16; lo half K chunks [0..7],[16..23],
//      hi half [8..15],[24..31]) ----
__device__ __forceinline__ v16h a_from_h(const _Float16* __restrict__ row, int k0, int hi) {
    const _Float16* p = row + k0 + (hi ? 8 : 0);
    v8h lo = *(const v8h*)(p);
    v8h hh = *(const v8h*)(p + 16);
    v16h r;
#pragma unroll
    for (int i = 0; i < 8; ++i) { r[i] = lo[i]; r[i + 8] = hh[i]; }
    return r;
}

// ---- B-operand loader (lane = N mod 16; lo half K chunks [0..7],[16..23],
//      hi half [8..15],[24..31]) ----
__device__ __forceinline__ v16h b_from_h(const _Float16* __restrict__ row, int k0, int hi) {
    const _Float16* p = row + k0 + (hi ? 8 : 0);
    v8h lo = *(const v8h*)(p);
    v8h hh = *(const v8h*)(p + 16);
    v16h r;
#pragma unroll
    for (int i = 0; i < 8; ++i) { r[i] = lo[i]; r[i + 8] = hh[i]; }
    return r;
}

// ============================================================================
// Kernel 0: one-time f32 -> f16 conversion (8 elements / thread).
// ============================================================================
__global__ void __launch_bounds__(256)
cvt_kernel(const float* __restrict__ src, _Float16* __restrict__ dst, int n) {
    const int i = (blockIdx.x * 256 + threadIdx.x) * 8;
    if (i + 7 < n) {
        v4f a = *(const v4f*)(src + i);
        v4f b = *(const v4f*)(src + i + 4);
        v8h o;
#pragma unroll
        for (int j = 0; j < 4; ++j) {
            o[j]     = (_Float16)a[j];
            o[j + 4] = (_Float16)b[j];
        }
        *(v8h*)(dst + i) = o;
    }
}

// ============================================================================
// Kernel 1: X16[4096,1024] @ W16^T + bias -> f16, rearranged per-head.
// mode 0: out[((b*H+h)*S + s)*HD + d]   (Q, K)
// mode 1: out[((b*H+h)*HD + d)*S + s]   (V, transposed for P@V B-operand)
// One wave computes a 32x64 output block: 2 A-tiles x 4 B-tiles -> 8 WMMAs
// per 32-wide K step (4 A-loads + 8 B-loads per 8 WMMAs).
// ============================================================================
__global__ void __launch_bounds__(256)
proj_kernel(const _Float16* __restrict__ X16, const _Float16* __restrict__ W16,
            const float* __restrict__ bias, _Float16* __restrict__ out, int mode) {
    const int lane = threadIdx.x & 31;
    const int ln   = lane & 15;
    const int hi   = lane >> 4;
    const int w    = blockIdx.x * (blockDim.x >> 5) + (threadIdx.x >> 5);
    const int nbase = (w & 15) * 64;   // 16 strips of 64 cols
    const int mbase = (w >> 4) * 32;   // 128 tiles of 32 rows

    const _Float16* arow0 = X16 + (size_t)(mbase +      ln) * D_;
    const _Float16* arow1 = X16 + (size_t)(mbase + 16 + ln) * D_;
    const _Float16* b0r   = W16 + (size_t)(nbase +  0 + ln) * D_;
    const _Float16* b1r   = W16 + (size_t)(nbase + 16 + ln) * D_;
    const _Float16* b2r   = W16 + (size_t)(nbase + 32 + ln) * D_;
    const _Float16* b3r   = W16 + (size_t)(nbase + 48 + ln) * D_;

    v8f acc[2][4];
#pragma unroll
    for (int r = 0; r < 2; ++r)
#pragma unroll
        for (int c = 0; c < 4; ++c) acc[r][c] = {};

    for (int k0 = 0; k0 < D_; k0 += 32) {
        const v16h a0 = a_from_h(arow0, k0, hi);
        const v16h a1 = a_from_h(arow1, k0, hi);
        const v16h b0 = b_from_h(b0r, k0, hi);
        const v16h b1 = b_from_h(b1r, k0, hi);
        const v16h b2 = b_from_h(b2r, k0, hi);
        const v16h b3 = b_from_h(b3r, k0, hi);
        acc[0][0] = wmma16(a0, b0, acc[0][0]);
        acc[0][1] = wmma16(a0, b1, acc[0][1]);
        acc[0][2] = wmma16(a0, b2, acc[0][2]);
        acc[0][3] = wmma16(a0, b3, acc[0][3]);
        acc[1][0] = wmma16(a1, b0, acc[1][0]);
        acc[1][1] = wmma16(a1, b1, acc[1][1]);
        acc[1][2] = wmma16(a1, b2, acc[1][2]);
        acc[1][3] = wmma16(a1, b3, acc[1][3]);
    }

#pragma unroll
    for (int c = 0; c < 4; ++c) {
        const int n  = nbase + c * 16 + ln;
        const float bn = bias[n];
        const int hh = n >> 6;     // head
        const int d  = n & 63;     // within-head dim
#pragma unroll
        for (int r = 0; r < 2; ++r) {
#pragma unroll
            for (int j = 0; j < 8; ++j) {
                const int row = mbase + r * 16 + j + hi * 8;   // b*S + s
                const int b   = row >> 11;
                const int s   = row & (S_ - 1);
                const float val = acc[r][c][j] + bn;
                size_t off;
                if (mode == 0) off = (((size_t)(b * H_ + hh)) * S_  + s) * HD_ + d;
                else           off = (((size_t)(b * H_ + hh)) * HD_ + d) * S_  + s;
                out[off] = (_Float16)val;
            }
        }
    }
}

// ============================================================================
// Kernel 2: causal flash attention per head.
// grid = (B*H, S/64); block = 128 (4 waves); each wave owns 16 query rows.
// Key blocks of 32: 4 WMMAs for scores, online softmax in C-layout registers,
// P re-laid to A-layout via 1KB per-wave LDS, 4 WMMAs for P@V.
// ============================================================================
__global__ void __launch_bounds__(128)
flash_kernel(const _Float16* __restrict__ Qh, const _Float16* __restrict__ Kh,
             const _Float16* __restrict__ Vt, _Float16* __restrict__ Oi) {
    __shared__ __align__(16) _Float16 pbuf[4][16][32];

    const int lane = threadIdx.x & 31;
    const int ln   = lane & 15;
    const int hi   = lane >> 4;
    const int wave = threadIdx.x >> 5;
    const int bh   = blockIdx.x;          // b*H + h
    const int b    = bh >> 4;
    const int h    = bh & 15;
    const int qb   = blockIdx.y * 64 + wave * 16;

    const _Float16* qrow = Qh + ((size_t)bh * S_ + qb + ln) * HD_;
    const v16h aq0 = a_from_h(qrow, 0,  hi);
    const v16h aq1 = a_from_h(qrow, 32, hi);

    const _Float16* vbase = Vt + (size_t)bh * HD_ * S_;
    const _Float16* v0r = vbase + (size_t)( 0 + ln) * S_;
    const _Float16* v1r = vbase + (size_t)(16 + ln) * S_;
    const _Float16* v2r = vbase + (size_t)(32 + ln) * S_;
    const _Float16* v3r = vbase + (size_t)(48 + ln) * S_;

    float rm[8], rl[8];
    v8f o0 = {}, o1 = {}, o2 = {}, o3 = {};
#pragma unroll
    for (int j = 0; j < 8; ++j) { rm[j] = -1e30f; rl[j] = 0.0f; }

    for (int t0 = 0; t0 < qb + 16; t0 += 32) {
        // ---- scores: S = (Q @ K^T) / 8 for keys [t0, t0+32) ----
        const _Float16* krow0 = Kh + ((size_t)bh * S_ + t0 + ln) * HD_;
        const _Float16* krow1 = krow0 + (size_t)16 * HD_;
        v8f s0 = {}, s1 = {};
        s0 = wmma16(aq0, b_from_h(krow0, 0,  hi), s0);
        s0 = wmma16(aq1, b_from_h(krow0, 32, hi), s0);
        s1 = wmma16(aq0, b_from_h(krow1, 0,  hi), s1);
        s1 = wmma16(aq1, b_from_h(krow1, 32, hi), s1);

        const bool need_mask = (t0 + 31 > qb);
#pragma unroll
        for (int j = 0; j < 8; ++j) {
            float r0 = s0[j] * 0.125f;     // 1/sqrt(HD)
            float r1 = s1[j] * 0.125f;
            if (need_mask) {
                const int rowg = qb + j + hi * 8;
                if (t0 + ln      > rowg) r0 = -1e30f;
                if (t0 + 16 + ln > rowg) r1 = -1e30f;
            }
            s0[j] = r0; s1[j] = r1;
        }

        // ---- online softmax (per output row = per accumulator slot j) ----
#pragma unroll
        for (int j = 0; j < 8; ++j) {
            float mj = fmaxf(s0[j], s1[j]);
#pragma unroll
            for (int off = 1; off < 16; off <<= 1)
                mj = fmaxf(mj, __shfl_xor(mj, off, 32));
            const float mn    = fmaxf(rm[j], mj);
            const float alpha = __expf(rm[j] - mn);
            rm[j] = mn;
            const float p0 = __expf(s0[j] - mn);
            const float p1 = __expf(s1[j] - mn);
            float sum = p0 + p1;
#pragma unroll
            for (int off = 1; off < 16; off <<= 1)
                sum += __shfl_xor(sum, off, 32);
            rl[j] = rl[j] * alpha + sum;
            o0[j] *= alpha; o1[j] *= alpha; o2[j] *= alpha; o3[j] *= alpha;
            // C-layout -> row-major f16 in per-wave LDS tile
            const int m = j + hi * 8;
            pbuf[wave][m][ln]      = (_Float16)p0;
            pbuf[wave][m][16 + ln] = (_Float16)p1;
        }

        // ---- P (A-layout from LDS) @ V-block -> accumulate O ----
        const v16h ap = a_from_h(&pbuf[wave][ln][0], 0, hi);
        o0 = wmma16(ap, b_from_h(v0r, t0, hi), o0);
        o1 = wmma16(ap, b_from_h(v1r, t0, hi), o1);
        o2 = wmma16(ap, b_from_h(v2r, t0, hi), o2);
        o3 = wmma16(ap, b_from_h(v3r, t0, hi), o3);
    }

    // ---- normalize and store f16 O tile into [B,S,D] layout ----
#pragma unroll
    for (int j = 0; j < 8; ++j) {
        const float inv = 1.0f / rl[j];
        const int s = qb + j + hi * 8;
        const size_t base = ((size_t)b * S_ + s) * D_ + h * HD_;
        Oi[base +  0 + ln] = (_Float16)(o0[j] * inv);
        Oi[base + 16 + ln] = (_Float16)(o1[j] * inv);
        Oi[base + 32 + ln] = (_Float16)(o2[j] * inv);
        Oi[base + 48 + ln] = (_Float16)(o3[j] * inv);
    }
}

// ============================================================================
// Kernel 3: Oi[4096,1024] (f16) @ Wo16^T + bo -> f32 out [B,S,D]
// Same 32x64 per-wave blocking as proj_kernel.
// ============================================================================
__global__ void __launch_bounds__(256)
oproj_kernel(const _Float16* __restrict__ Oi, const _Float16* __restrict__ W16,
             const float* __restrict__ bias, float* __restrict__ out) {
    const int lane = threadIdx.x & 31;
    const int ln   = lane & 15;
    const int hi   = lane >> 4;
    const int w    = blockIdx.x * (blockDim.x >> 5) + (threadIdx.x >> 5);
    const int nbase = (w & 15) * 64;
    const int mbase = (w >> 4) * 32;

    const _Float16* arow0 = Oi + (size_t)(mbase +      ln) * D_;
    const _Float16* arow1 = Oi + (size_t)(mbase + 16 + ln) * D_;
    const _Float16* b0r   = W16 + (size_t)(nbase +  0 + ln) * D_;
    const _Float16* b1r   = W16 + (size_t)(nbase + 16 + ln) * D_;
    const _Float16* b2r   = W16 + (size_t)(nbase + 32 + ln) * D_;
    const _Float16* b3r   = W16 + (size_t)(nbase + 48 + ln) * D_;

    v8f acc[2][4];
#pragma unroll
    for (int r = 0; r < 2; ++r)
#pragma unroll
        for (int c = 0; c < 4; ++c) acc[r][c] = {};

    for (int k0 = 0; k0 < D_; k0 += 32) {
        const v16h a0 = a_from_h(arow0, k0, hi);
        const v16h a1 = a_from_h(arow1, k0, hi);
        const v16h b0 = b_from_h(b0r, k0, hi);
        const v16h b1 = b_from_h(b1r, k0, hi);
        const v16h b2 = b_from_h(b2r, k0, hi);
        const v16h b3 = b_from_h(b3r, k0, hi);
        acc[0][0] = wmma16(a0, b0, acc[0][0]);
        acc[0][1] = wmma16(a0, b1, acc[0][1]);
        acc[0][2] = wmma16(a0, b2, acc[0][2]);
        acc[0][3] = wmma16(a0, b3, acc[0][3]);
        acc[1][0] = wmma16(a1, b0, acc[1][0]);
        acc[1][1] = wmma16(a1, b1, acc[1][1]);
        acc[1][2] = wmma16(a1, b2, acc[1][2]);
        acc[1][3] = wmma16(a1, b3, acc[1][3]);
    }

#pragma unroll
    for (int c = 0; c < 4; ++c) {
        const int n = nbase + c * 16 + ln;
        const float bn = bias[n];
#pragma unroll
        for (int r = 0; r < 2; ++r) {
#pragma unroll
            for (int j = 0; j < 8; ++j) {
                const int row = mbase + r * 16 + j + hi * 8;
                out[(size_t)row * D_ + n] = acc[r][c][j] + bn;
            }
        }
    }
}

// ============================================================================
extern "C" void kernel_launch(void* const* d_in, const int* in_sizes, int n_in,
                              void* d_out, int out_size, void* d_ws, size_t ws_size,
                              hipStream_t stream) {
    const float* q  = (const float*)d_in[0];
    const float* k  = (const float*)d_in[1];
    const float* v  = (const float*)d_in[2];
    const float* Wq = (const float*)d_in[3];
    const float* bq = (const float*)d_in[4];
    const float* Wk = (const float*)d_in[5];
    const float* bk = (const float*)d_in[6];
    const float* Wv = (const float*)d_in[7];
    const float* bv = (const float*)d_in[8];
    const float* Wo = (const float*)d_in[9];
    const float* bo = (const float*)d_in[10];
    float* out = (float*)d_out;

    _Float16* ws = (_Float16*)d_ws;
    const size_t NW = (size_t)D_ * D_;                 // 1M elements per weight
    const size_t NX = (size_t)B_ * S_ * D_;            // 4M elements per activation
    // Workspace layout (halfs):
    //  [0,4M):   Wq16, Wk16, Wv16, Wo16
    //  [4M,16M): Qh, Kh, Vt  ([B,H,S,HD] / [B,H,HD,S])
    //  [16M,28M): Xq16, Xk16, Xv16  (Xq16 region reused as Oi after projections)
    _Float16* Wq16 = ws;
    _Float16* Wk16 = ws + NW;
    _Float16* Wv16 = ws + 2 * NW;
    _Float16* Wo16 = ws + 3 * NW;
    _Float16* Qh   = ws + 4 * NW;
    _Float16* Kh   = Qh + NX;
    _Float16* Vt   = Qh + 2 * NX;
    _Float16* Xq16 = Qh + 3 * NX;
    _Float16* Xk16 = Qh + 4 * NX;
    _Float16* Xv16 = Qh + 5 * NX;
    _Float16* Oi   = Xq16;             // alias: q's f16 copy is dead post-proj
    (void)ws_size;                     // requires >= 56 MB

    // Phase 0: one-time f32 -> f16 conversions.
    cvt_kernel<<<dim3((int)(NW / 2048)), dim3(256), 0, stream>>>(Wq, Wq16, (int)NW);
    cvt_kernel<<<dim3((int)(NW / 2048)), dim3(256), 0, stream>>>(Wk, Wk16, (int)NW);
    cvt_kernel<<<dim3((int)(NW / 2048)), dim3(256), 0, stream>>>(Wv, Wv16, (int)NW);
    cvt_kernel<<<dim3((int)(NW / 2048)), dim3(256), 0, stream>>>(Wo, Wo16, (int)NW);
    cvt_kernel<<<dim3((int)(NX / 2048)), dim3(256), 0, stream>>>(q, Xq16, (int)NX);
    cvt_kernel<<<dim3((int)(NX / 2048)), dim3(256), 0, stream>>>(k, Xk16, (int)NX);
    cvt_kernel<<<dim3((int)(NX / 2048)), dim3(256), 0, stream>>>(v, Xv16, (int)NX);

    // Phase 1: projections. 2048 waves: 128 M-tiles(32) x 16 N-strips(64).
    const dim3 pgrid(256), pblock(256);
    proj_kernel<<<pgrid, pblock, 0, stream>>>(Xq16, Wq16, bq, Qh, 0);
    proj_kernel<<<pgrid, pblock, 0, stream>>>(Xk16, Wk16, bk, Kh, 0);
    proj_kernel<<<pgrid, pblock, 0, stream>>>(Xv16, Wv16, bv, Vt, 1);

    // Phase 2: causal flash attention (writes Oi, aliasing Xq16).
    flash_kernel<<<dim3(B_ * H_, S_ / 64), dim3(128), 0, stream>>>(Qh, Kh, Vt, Oi);

    // Phase 3: output projection -> f32.
    oproj_kernel<<<pgrid, pblock, 0, stream>>>(Oi, Wo16, bo, out);
}
